// MixLinear_GEMM_3298534884151
// MI455X (gfx1250) — 8-bit matrix operand order repaired and checked by reading, compiled, not run on hardware
//
#include <hip/hip_runtime.h>
#include <hip/hip_bf16.h>
#include <stdint.h>

#define FP8_MAX 448.0f
#define TILEK   128
#define BSTRIDE 144

typedef __attribute__((ext_vector_type(16))) int      v16i;
typedef __attribute__((ext_vector_type(8)))  float    v8f;
typedef __attribute__((ext_vector_type(2)))  float    v2f;
typedef __attribute__((ext_vector_type(4)))  unsigned u32x4;
typedef __attribute__((ext_vector_type(8)))  int      i32x8;
typedef __attribute__((ext_vector_type(4)))  int      i32x4;

#if __has_builtin(__builtin_amdgcn_tensor_load_to_lds)
#define HAVE_TDM 1
#else
#define HAVE_TDM 0
#endif

__device__ __forceinline__ unsigned f32_to_e4m3_sw(float x) {
    unsigned bits = __float_as_uint(x);
    unsigned sign = (bits >> 24) & 0x80u;
    float ax = fabsf(x);
    if (!(ax > 0.0f)) return sign;
    if (ax > FP8_MAX) ax = FP8_MAX;
    if (ax < 0.015625f) {
        unsigned m = (unsigned)(ax * 512.0f + 0.5f);
        return sign | m;
    }
    unsigned u = __float_as_uint(ax);
    unsigned q = (((u >> 23) - 120u) << 3) | ((u >> 20) & 7u);
    q += (u >> 19) & 1u;
    if (q > 0x7Eu) q = 0x7Eu;
    return sign | q;
}

__device__ __forceinline__ unsigned pack4_e4m3(float a, float b, float c, float d) {
#if __has_builtin(__builtin_amdgcn_cvt_pk_fp8_f32)
    a = fminf(fmaxf(a, -FP8_MAX), FP8_MAX);
    b = fminf(fmaxf(b, -FP8_MAX), FP8_MAX);
    c = fminf(fmaxf(c, -FP8_MAX), FP8_MAX);
    d = fminf(fmaxf(d, -FP8_MAX), FP8_MAX);
    int v = __builtin_amdgcn_cvt_pk_fp8_f32(a, b, 0, false);
    v     = __builtin_amdgcn_cvt_pk_fp8_f32(c, d, v, true);
    return (unsigned)v;
#else
    return f32_to_e4m3_sw(a) | (f32_to_e4m3_sw(b) << 8) |
           (f32_to_e4m3_sw(c) << 16) | (f32_to_e4m3_sw(d) << 24);
#endif
}

__global__ void k_zero_hdr(unsigned* hdr) {
    if (threadIdx.x < 16) hdr[threadIdx.x] = 0u;
}

__global__ void k_absmax_x(const float* __restrict__ x, long long n, unsigned* hdr) {
    long long i      = (long long)blockIdx.x * blockDim.x + threadIdx.x;
    long long stride = (long long)gridDim.x * blockDim.x;
    float m = 0.0f;
    for (; i < n; i += stride) m = fmaxf(m, fabsf(x[i]));
    #pragma unroll
    for (int off = 16; off > 0; off >>= 1) m = fmaxf(m, __shfl_xor(m, off, 32));
    if ((threadIdx.x & 31) == 0) atomicMax(hdr + 0, __float_as_uint(m));
}

__global__ void k_absmax_w(const int* __restrict__ qw, const float* __restrict__ qs,
                           int Kp, unsigned* hdr) {
    int c = blockIdx.x * blockDim.x + threadIdx.x;
    int n = blockIdx.y;
    float m = 0.0f;
    if (c < Kp) {
        unsigned w = (unsigned)qw[(long long)n * Kp + c];
        int mx = 0;
        #pragma unroll
        for (int j = 0; j < 8; ++j) {
            int nib = (int)((w >> (4 * j)) & 0xFu) - 8;
            int d = nib < 0 ? -nib : nib;
            mx = d > mx ? d : mx;
        }
        m = (float)mx * qs[(long long)n * (Kp >> 4) + (c >> 4)];
    }
    #pragma unroll
    for (int off = 16; off > 0; off >>= 1) m = fmaxf(m, __shfl_xor(m, off, 32));
    if ((threadIdx.x & 31) == 0) atomicMax(hdr + 1, __float_as_uint(m));
}

__global__ void k_finalize(unsigned* hdr) {
    if (threadIdx.x == 0) {
        float sx = __uint_as_float(hdr[0]) / FP8_MAX;
        float sw = __uint_as_float(hdr[1]) / FP8_MAX;
        float* f = (float*)hdr;
        f[4] = sx;
        f[5] = sw;
        f[6] = sx > 0.0f ? 1.0f / sx : 0.0f;
        f[7] = sw > 0.0f ? 1.0f / sw : 0.0f;
    }
}

__global__ void k_quant_x(const float* __restrict__ x, unsigned* __restrict__ xq,
                          const float* __restrict__ hdrf, long long n4) {
    long long i = (long long)blockIdx.x * blockDim.x + threadIdx.x;
    if (i >= n4) return;
    float rs = hdrf[6];
    float4 v = ((const float4*)x)[i];
    xq[i] = pack4_e4m3(v.x * rs, v.y * rs, v.z * rs, v.w * rs);
}

__global__ void k_quant_w(const int* __restrict__ qw, const float* __restrict__ qs,
                          unsigned* __restrict__ wq, const float* __restrict__ hdrf,
                          int Kp) {
    int c = blockIdx.x * blockDim.x + threadIdx.x;
    int n = blockIdx.y;
    if (c >= Kp) return;
    float rs = hdrf[7] * qs[(long long)n * (Kp >> 4) + (c >> 4)];
    unsigned w = (unsigned)qw[(long long)n * Kp + c];
    float f[8];
    #pragma unroll
    for (int j = 0; j < 8; ++j)
        f[j] = (float)((int)((w >> (4 * j)) & 0xFu) - 8) * rs;
    long long o = ((long long)n * Kp + c) * 2;
    wq[o + 0] = pack4_e4m3(f[0], f[1], f[2], f[3]);
    wq[o + 1] = pack4_e4m3(f[4], f[5], f[6], f[7]);
}

__global__ void k_gather_x(const float* __restrict__ x, const int* __restrict__ ind,
                           float* __restrict__ xg, int K, int FP, long long total) {
    long long i = (long long)blockIdx.x * blockDim.x + threadIdx.x;
    if (i >= total) return;
    long long m = i / FP;
    int f = (int)(i - m * FP);
    xg[i] = x[m * K + ind[f]];
}

__device__ __forceinline__ void tdm_load_tile(const unsigned char* gsrc,
                                              unsigned lds_off, int K, int N) {
#if HAVE_TDM
    unsigned long long ga = (unsigned long long)gsrc;
    u32x4 g0 = {0u, 0u, 0u, 0u};
    g0[0] = 1u;
    g0[1] = lds_off;
    g0[2] = (unsigned)(ga & 0xFFFFFFFFu);
    g0[3] = (unsigned)((ga >> 32) & 0x01FFFFFFu)
          | (2u << 30);
    i32x8 g1 = {0, 0, 0, 0, 0, 0, 0, 0};
    g1[0] = (int)((1u << 20) | (4u << 22) | (3u << 25));
    g1[1] = (int)(((unsigned)K & 0xFFFFu) << 16);
    g1[2] = (int)((((unsigned)K >> 16) & 0xFFFFu) |
                  (((unsigned)N & 0xFFFFu) << 16));
    g1[3] = (int)(((unsigned)N >> 16) | (128u << 16));
    g1[4] = (int)128u;
    g1[5] = (int)K;
    g1[6] = 0;
    g1[7] = 0;
    i32x4 z4 = {0, 0, 0, 0};
#if defined(__clang_major__) && __clang_major__ >= 23
    i32x8 z8 = {0, 0, 0, 0, 0, 0, 0, 0};
    __builtin_amdgcn_tensor_load_to_lds(g0, g1, z4, z4, z8, 0);
#else
    __builtin_amdgcn_tensor_load_to_lds(g0, g1, z4, z4, 0);
#endif
#else
    (void)gsrc; (void)lds_off; (void)K; (void)N;
#endif
}

__global__ __launch_bounds__(256)
void k_gemm_fp8(const unsigned char* __restrict__ Xq,
                const unsigned char* __restrict__ Wq,
                const float* __restrict__ bias,
                const float* __restrict__ hdrf,
                float* __restrict__ y, int M, int N, int K) {
    __shared__ unsigned char sB[2][128 * BSTRIDE];

    const int lane = threadIdx.x & 31;
    const int wave = threadIdx.x >> 5;
    const int m0   = blockIdx.y * 128 + wave * 16;
    const int n0   = blockIdx.x * 128;
    const int row  = lane & 15;
    const int hi   = lane >> 4;
    const int ka   = hi * 16;
    const int kb   = hi * 16;

    unsigned ldsbase[2];
    ldsbase[0] = (unsigned)(unsigned long long)(const void*)&sB[0][0];
    ldsbase[1] = (unsigned)(unsigned long long)(const void*)&sB[1][0];

    v8f acc[8];
    #pragma unroll
    for (int j = 0; j < 8; ++j)
        #pragma unroll
        for (int r = 0; r < 8; ++r) acc[j][r] = 0.0f;

    const unsigned char* aRow  = Xq + (long long)(m0 + row) * K;
    const unsigned char* bBase = Wq + (long long)n0 * K;

    auto stage = [&](int kt, int buf) {
#if HAVE_TDM
        if (wave == 0) tdm_load_tile(bBase + kt, ldsbase[buf], K, N);
#else
        int r = threadIdx.x >> 1, h = (threadIdx.x & 1) * 64;
        const uint4* src = (const uint4*)(bBase + (long long)r * K + kt + h);
        uint4* dst = (uint4*)(&sB[buf][r * BSTRIDE + h]);
        #pragma unroll
        for (int q = 0; q < 4; ++q) dst[q] = src[q];
#endif
    };

    auto load_bfrag = [&](v16i& b, int buf, int j) {
        const unsigned char* p = &sB[buf][(j * 16 + row) * BSTRIDE + kb];
        #pragma unroll
        for (int g = 0; g < 4; ++g) {
            uint4 t = *(const uint4*)(p + 32 * g);
            b[4 * g]     = (int)t.x;
            b[4 * g + 1] = (int)t.y;
            b[4 * g + 2] = (int)t.z;
            b[4 * g + 3] = (int)t.w;
        }
    };

    stage(0, 0);
#if HAVE_TDM
    __builtin_amdgcn_s_wait_tensorcnt(0);
#endif
    __syncthreads();

    int cur = 0;
    for (int kk = 0; kk < K; kk += TILEK) {
        if (kk + TILEK < K) {
            stage(kk + TILEK, cur ^ 1);
            __builtin_prefetch(aRow + kk + TILEK, 0, 1);
        }

        v16i a;
        #pragma unroll
        for (int p = 0; p < 8; ++p) {
            uint2 t = *(const uint2*)(aRow + kk + ka + 32 * (p >> 1) + 8 * (p & 1));
            a[2 * p]     = (int)t.x;
            a[2 * p + 1] = (int)t.y;
        }

        v16i b0, b1;
        load_bfrag(b0, cur, 0);
        #pragma unroll
        for (int j = 0; j < 8; ++j) {
            if (j + 1 < 8) load_bfrag(((j & 1) ? b0 : b1), cur, j + 1);
            acc[j] = __builtin_amdgcn_wmma_f32_16x16x128_fp8_fp8(
                a, ((j & 1) ? b1 : b0), (short)0, acc[j], false, false);
        }

#if HAVE_TDM
        __builtin_amdgcn_s_wait_tensorcnt(0);
#endif
        __syncthreads();
        cur ^= 1;
    }

    const float s = hdrf[4] * hdrf[5];
    const int mB  = m0 + 8 * hi;
    const int nB  = n0 + (lane & 15);
    #pragma unroll
    for (int j = 0; j < 8; ++j) {
        const int n = nB + j * 16;
        const float bv = bias[n];
        #pragma unroll
        for (int r = 0; r < 8; ++r)
            y[(long long)(mB + r) * N + n] = acc[j][r] * s + bv;
    }
}

__global__ __launch_bounds__(256)
void k_correction(const float* __restrict__ xg, const float* __restrict__ wc,
                  float* __restrict__ y, int M, int N, int FP) {
    const int lane = threadIdx.x & 31;
    const int wave = threadIdx.x >> 5;
    const int m0   = blockIdx.y * 128 + wave * 16;
    const int n0   = blockIdx.x * 16;
    const int row  = lane & 15;
    const int hi   = lane >> 4;

    v8f acc;
    #pragma unroll
    for (int r = 0; r < 8; ++r) acc[r] = 0.0f;

    const float* aR = xg + (long long)(m0 + row) * FP + hi * 2;
    const float* bR = wc + (long long)(n0 + row) * FP + hi * 2;

#if __has_builtin(__builtin_amdgcn_wmma_f32_16x16x4_f32)
    for (int k = 0; k < FP; k += 4) {
        v2f a, b;
        a[0] = aR[k]; a[1] = aR[k + 1];
        b[0] = bR[k]; b[1] = bR[k + 1];
        acc = __builtin_amdgcn_wmma_f32_16x16x4_f32(false, a, false, b,
                                                    (short)0, acc, false, false);
    }
#else
    #pragma unroll 1
    for (int k = 0; k < FP; ++k) {
        const float bvv = wc[(long long)(n0 + row) * FP + k];
        (void)bvv;
        #pragma unroll
        for (int r = 0; r < 8; ++r)
            acc[r] += xg[(long long)(m0 + 8 * hi + r) * FP + k] *
                      wc[(long long)(n0 + (lane & 15)) * FP + k];
    }
#endif

    const int mB = m0 + 8 * hi;
    const int n  = n0 + (lane & 15);
    #pragma unroll
    for (int r = 0; r < 8; ++r)
        y[(long long)(mB + r) * N + n] += acc[r];
}

extern "C" void kernel_launch(void* const* d_in, const int* in_sizes, int n_in,
                              void* d_out, int out_size, void* d_ws, size_t ws_size,
                              hipStream_t stream) {
    const float* x    = (const float*)d_in[0];
    const int*   qw   = (const int*)d_in[1];
    const float* qs   = (const float*)d_in[2];
    const float* wc   = (const float*)d_in[3];
    const int*   ind  = (const int*)d_in[4];
    const float* bias = (const float*)d_in[5];
    float*       y    = (float*)d_out;

    const int FP = in_sizes[4];
    const int N  = in_sizes[5];
    const int Kp = in_sizes[1] / N;
    const int K  = Kp * 8;
    const int M  = in_sizes[0] / K;

    char* ws = (char*)d_ws;
    unsigned*      hdr  = (unsigned*)ws;
    float*         hdrf = (float*)ws;
    unsigned char* Xq   = (unsigned char*)(ws + 256);
    unsigned char* Wq   = Xq + (size_t)M * K;
    float*         xg   = (float*)(Wq + (size_t)N * K);

    k_zero_hdr<<<1, 64, 0, stream>>>(hdr);
    k_absmax_x<<<2048, 256, 0, stream>>>(x, (long long)M * K, hdr);
    dim3 gw((Kp + 255) / 256, N);
    k_absmax_w<<<gw, 256, 0, stream>>>(qw, qs, Kp, hdr);
    k_finalize<<<1, 32, 0, stream>>>(hdr);
    long long n4 = (long long)M * K / 4;
    k_quant_x<<<(unsigned)((n4 + 255) / 256), 256, 0, stream>>>(x, (unsigned*)Xq, hdrf, n4);
    k_quant_w<<<gw, 256, 0, stream>>>(qw, qs, (unsigned*)Wq, hdrf, Kp);
    long long tg = (long long)M * FP;
    k_gather_x<<<(unsigned)((tg + 255) / 256), 256, 0, stream>>>(x, ind, xg, K, FP, tg);
    dim3 gg(N / 128, M / 128);
    k_gemm_fp8<<<gg, 256, 0, stream>>>(Xq, Wq, bias, hdrf, y, M, N, K);
    dim3 gc(N / 16, M / 128);
    k_correction<<<gc, 256, 0, stream>>>(xg, wc, y, M, N, FP);
}
